// SymbolicSequenceRNN_79302276153627
// MI455X (gfx1250) — matrix operand order repaired and checked by reading, not compiled, not run on hardware
//
#include <hip/hip_runtime.h>
#include <hip/hip_bf16.h>

// Problem dims
#define B_ 4
#define S_ 512
#define H_ 256
#define G_ 768    // 3*H
#define V_ 32000

typedef __attribute__((ext_vector_type(16))) __bf16 v16bf;
typedef __attribute__((ext_vector_type(8)))  float  v8f;
typedef __attribute__((ext_vector_type(4)))  int    v4i;

union Frag {
  v16bf v;
  unsigned short u[16];
  uint4 q[2];
};

// Async global->LDS path (CDNA5): guard on builtin availability.
#if defined(__has_builtin)
#if __has_builtin(__builtin_amdgcn_global_load_async_to_lds_b128) && \
    __has_builtin(__builtin_amdgcn_s_wait_asynccnt)
#define HAVE_ASYNC_LDS 1
#endif
#endif
#ifndef HAVE_ASYNC_LDS
#define HAVE_ASYNC_LDS 0
#endif

// round-to-nearest-even f32 -> bf16 (bit pattern)
__device__ __forceinline__ unsigned short f2bf(float x) {
  unsigned int t = __float_as_uint(x);
  t += 0x7fffu + ((t >> 16) & 1u);
  return (unsigned short)(t >> 16);
}

__device__ __forceinline__ v8f wmma_bf16(v16bf a, v16bf b, v8f c) {
  // D = A(16x32 bf16) x B(32x16 bf16) + C(16x16 f32)
  return __builtin_amdgcn_wmma_f32_16x16x32_bf16(
      /*neg_a=*/false, a, /*neg_b=*/false, b,
      /*c_mod=*/(short)0, c, /*reuse_a=*/false, /*reuse_b=*/false);
}

// ---------------------------------------------------------------------------
// Kernel 0 (optional): W_out f32 -> bf16, so the big GEMM streams half bytes.
// ---------------------------------------------------------------------------
__global__ __launch_bounds__(256)
void k_cvt_wout(const float* __restrict__ Wout, unsigned short* __restrict__ Wb) {
  const int idx = blockIdx.x * 256 + threadIdx.x;   // [0, V*H/4)
  const float4 f = ((const float4*)Wout)[idx];
  ushort4 o;
  o.x = f2bf(f.x); o.y = f2bf(f.y); o.z = f2bf(f.z); o.w = f2bf(f.w);
  ((ushort4*)Wb)[idx] = o;
}

// ---------------------------------------------------------------------------
// Kernel 1: xp[m][g] = dot(emb[x[m]], W_ih[g]) + b_ih[g]   (f32, exact-ish)
// ---------------------------------------------------------------------------
__global__ __launch_bounds__(256)
void k_xproj(const int* __restrict__ x, const float* __restrict__ emb,
             const float* __restrict__ Wih, const float* __restrict__ bih,
             float* __restrict__ xp) {
  const int idx = blockIdx.x * 256 + threadIdx.x;  // [0, 2048*768)
  const int m = idx / G_;
  const int g = idx % G_;
  const float4* e4 = (const float4*)(emb + (size_t)x[m] * H_);
  const float4* w4 = (const float4*)(Wih + (size_t)g * H_);
  float acc = bih[g];
#pragma unroll 8
  for (int k = 0; k < H_ / 4; ++k) {
    const float4 a = e4[k];
    const float4 b = w4[k];
    acc += a.x * b.x + a.y * b.y + a.z * b.z + a.w * b.w;
  }
  xp[idx] = acc;
}

// ---------------------------------------------------------------------------
// Kernel 2: GRU recurrence. One workgroup, 512 threads = 16 waves.
// Wave w owns gate tiles {w, w+16, w+32} => hidden units [16w,16w+16) for all
// three gates. W_hh register-resident as bf16 B-fragments. Double-buffered h
// in LDS (one barrier/step). A-fragment LDS loads software-pipelined so the
// ds latency overlaps the previous chunk's WMMAs on the serial chain.
// ---------------------------------------------------------------------------
__global__ __launch_bounds__(512, 1)
void k_gru(const float* __restrict__ xp, const float* __restrict__ Whh,
           const float* __restrict__ bhh, unsigned short* __restrict__ hseq) {
  __shared__ __attribute__((aligned(16))) unsigned short hl[2][16 * H_];

  const int tid  = threadIdx.x;
  const int w    = tid >> 5;        // wave id [0,16)
  const int lane = tid & 31;
  const int kh   = lane >> 4;       // half-wave (K subset select)
  const int lr   = lane & 15;       // row/col within tile

  for (int i = tid; i < 2 * 16 * H_; i += 512) ((unsigned short*)hl)[i] = 0;

  // W_hh as register-resident bf16 B-fragments: B[k][n] = W_hh[n][k]
  v16bf Bw[3][8];
#pragma unroll
  for (int g = 0; g < 3; ++g) {
    const int row = g * H_ + w * 16 + lr;  // global W_hh row (gate unit)
#pragma unroll
    for (int kc = 0; kc < 8; ++kc) {
      const float* src = Whh + (size_t)row * H_ + kc * 32 + kh * 16;
      Frag f;
#pragma unroll
      for (int e = 0; e < 16; ++e) f.u[e] = f2bf(src[e]);
      Bw[g][kc] = f.v;
    }
  }

  const int j = w * 16 + lr;  // hidden unit owned by lanes 0..15
  float br = 0.f, bz = 0.f, bn = 0.f;
  float hcur[B_] = {0.f, 0.f, 0.f, 0.f};  // f32 master copy of h[b][j]
  if (lane < 16) {
    br = bhh[j];
    bz = bhh[H_ + j];
    bn = bhh[2 * H_ + j];
  }

  const int abase = lr * H_ + kh * 16;  // A-fragment base (per lane)

  __syncthreads();

  for (int t = 0; t < S_; ++t) {
    const unsigned short* rb = hl[t & 1];        // read buffer (h_t)
    unsigned short*       wb = hl[(t + 1) & 1];  // write buffer (h_{t+1})

    // xp loads are h-independent: issue early.
    float xr[B_], xz[B_], xn[B_];
    if (lane < 16) {
#pragma unroll
      for (int b = 0; b < B_; ++b) {
        const float* p = xp + (size_t)(b * S_ + t) * G_;
        xr[b] = p[j];
        xz[b] = p[H_ + j];
        xn[b] = p[2 * H_ + j];
      }
    }

    // hp = h @ W_hh^T via 24 bf16 WMMAs; A loads pipelined one chunk ahead.
    v8f cr = {}, cz = {}, cn = {};
    Frag a0, a1;
    a0.q[0] = *(const uint4*)&rb[abase];
    a0.q[1] = *(const uint4*)&rb[abase + 8];
#pragma unroll
    for (int kc = 0; kc < 8; ++kc) {
      Frag& cur = (kc & 1) ? a1 : a0;
      Frag& nxt = (kc & 1) ? a0 : a1;
      if (kc < 7) {  // prefetch chunk kc+1 before consuming chunk kc
        const int base = abase + (kc + 1) * 32;
        nxt.q[0] = *(const uint4*)&rb[base];
        nxt.q[1] = *(const uint4*)&rb[base + 8];
      }
      cr = wmma_bf16(cur.v, Bw[0][kc], cr);
      cz = wmma_bf16(cur.v, Bw[1][kc], cz);
      cn = wmma_bf16(cur.v, Bw[2][kc], cn);
    }

    if (lane < 16) {
#pragma unroll
      for (int b = 0; b < B_; ++b) {
        // C layout: element b (=M row b), lanes 0..15, N = lr -> unit j
        const float r = 1.f / (1.f + __expf(-(xr[b] + cr[b] + br)));
        const float z = 1.f / (1.f + __expf(-(xz[b] + cz[b] + bz)));
        const float n = tanhf(xn[b] + r * (cn[b] + bn));
        const float hnew = (1.f - z) * n + z * hcur[b];
        hcur[b] = hnew;
        const unsigned short hb = f2bf(hnew);
        wb[b * H_ + j] = hb;                              // disjoint from rb
        hseq[(size_t)(b * S_ + t) * H_ + j] = hb;         // bf16 A for GEMM
      }
    }
    __syncthreads();  // wb complete before next step reads it
  }
}

// ---------------------------------------------------------------------------
// Kernel 3: logits[m][n] = sum_k hseq[m][k]*W_out[n][k] + b_out[n], masked.
// Block = 256 threads (8 waves). The block's 128-row A panel (64 KB bf16) is
// staged once into LDS via async-to-LDS and shared by all waves. Wave w owns
// vocab tile nt = blockIdx.x*8+w with W_out register-resident, loops 8
// M-tiles reading A from LDS (pipelined).
// ---------------------------------------------------------------------------
template <bool BF16W>
__global__ __launch_bounds__(256)
void k_logits(const unsigned short* __restrict__ hseq,
              const void* __restrict__ WoutAny, const float* __restrict__ bout,
              const int* __restrict__ x, float* __restrict__ out) {
  __shared__ __attribute__((aligned(16))) unsigned short Apanel[128 * H_];  // 64 KB

  const int w    = threadIdx.x >> 5;
  const int lane = threadIdx.x & 31;
  const int kh   = lane >> 4;
  const int lr   = lane & 15;

  // ---- Stage the 64 KB contiguous A panel (rows blockIdx.y*128 .. +128) ----
  {
    const char* g = (const char*)(hseq + (size_t)blockIdx.y * 128 * H_);
#if HAVE_ASYNC_LDS
    char* l = (char*)Apanel;
#pragma unroll
    for (int it = 0; it < 16; ++it) {
      const int off = ((int)threadIdx.x + it * 256) * 16;
      __builtin_amdgcn_global_load_async_to_lds_b128(
          (__attribute__((address_space(1))) v4i*)(g + off),
          (__attribute__((address_space(3))) v4i*)(l + off),
          /*offset=*/0, /*cpol=*/0);
    }
    __builtin_amdgcn_s_wait_asynccnt(0);
#else
#pragma unroll
    for (int it = 0; it < 16; ++it) {
      const int off = ((int)threadIdx.x + it * 256) * 16;
      *(uint4*)((char*)Apanel + off) = *(const uint4*)(g + off);
    }
#endif
    __syncthreads();
  }

  const int nt = blockIdx.x * 8 + w;  // [0, 2000)
  const int n  = nt * 16 + lr;        // vocab column
  const float bo = bout[n];
  const bool nge3 = (n >= 3);
  const float NEG_INF = __int_as_float(0xff800000);

  // ---- Resident B fragments for this vocab tile: B[k][n] = W_out[n][k] ----
  v16bf Bw[8];
#pragma unroll
  for (int kc = 0; kc < 8; ++kc) {
    Frag f;
    if (BF16W) {
      const unsigned short* src =
          (const unsigned short*)WoutAny + (size_t)n * H_ + kc * 32 + kh * 16;
      f.q[0] = *(const uint4*)src;
      f.q[1] = *(const uint4*)(src + 8);
    } else {
      const float* src = (const float*)WoutAny + (size_t)n * H_ + kc * 32 + kh * 16;
#pragma unroll
      for (int e = 0; e < 16; ++e) f.u[e] = f2bf(src[e]);
    }
    Bw[kc] = f.v;
  }

  for (int i = 0; i < 8; ++i) {
    const int mt = blockIdx.y * 8 + i;  // [0, 128)
    const int abase = (i * 16 + lr) * H_ + kh * 16;  // panel-local A base
    v8f c = {};
    Frag a0, a1;
    a0.q[0] = *(const uint4*)&Apanel[abase];
    a0.q[1] = *(const uint4*)&Apanel[abase + 8];
#pragma unroll
    for (int kc = 0; kc < 8; ++kc) {
      Frag& cur = (kc & 1) ? a1 : a0;
      Frag& nxt = (kc & 1) ? a0 : a1;
      if (kc < 7) {
        const int base = abase + (kc + 1) * 32;
        nxt.q[0] = *(const uint4*)&Apanel[base];
        nxt.q[1] = *(const uint4*)&Apanel[base + 8];
      }
      c = wmma_bf16(cur.v, Bw[kc], c);
    }
    // C element v -> M = mt*16 + v + kh*8, N = n. Fuse bias + mask + store.
#pragma unroll
    for (int v = 0; v < 8; ++v) {
      const int m = mt * 16 + v + kh * 8;   // flattened (b, t)
      const int t = m & (S_ - 1);
      const int b = m >> 9;
      const bool masked = nge3 && (t > 0) && (x[b * S_ + t - 1] == 0);
      out[(size_t)m * V_ + n] = masked ? NEG_INF : (c[v] + bo);
    }
  }
}

// ---------------------------------------------------------------------------
extern "C" void kernel_launch(void* const* d_in, const int* in_sizes, int n_in,
                              void* d_out, int out_size, void* d_ws, size_t ws_size,
                              hipStream_t stream) {
  const int*   x    = (const int*)d_in[0];
  const float* emb  = (const float*)d_in[1];
  const float* Wih  = (const float*)d_in[2];
  const float* Whh  = (const float*)d_in[3];
  const float* bih  = (const float*)d_in[4];
  const float* bhh  = (const float*)d_in[5];
  const float* Wout = (const float*)d_in[6];
  const float* bout = (const float*)d_in[7];
  float* out = (float*)d_out;

  // Workspace layout: xp f32 (6.29 MB) | hseq bf16 (1 MB) | [Wout bf16 16.4 MB]
  const size_t xp_bytes   = (size_t)B_ * S_ * G_ * sizeof(float);
  const size_t hseq_bytes = (size_t)B_ * S_ * H_ * sizeof(unsigned short);
  const size_t wout_bytes = (size_t)V_ * H_ * sizeof(unsigned short);

  float* xp = (float*)d_ws;
  unsigned short* hseq = (unsigned short*)((char*)d_ws + xp_bytes);
  unsigned short* woutb =
      (unsigned short*)((char*)d_ws + xp_bytes + hseq_bytes);
  const bool use_bf16_w = (ws_size >= xp_bytes + hseq_bytes + wout_bytes);

  // 1) embedding gather + input projections (hoisted out of the scan)
  k_xproj<<<dim3((B_ * S_ * G_) / 256), dim3(256), 0, stream>>>(x, emb, Wih, bih, xp);

  // 1b) optional one-time W_out -> bf16 conversion (halves GEMM read traffic)
  if (use_bf16_w) {
    k_cvt_wout<<<dim3((V_ * H_ / 4) / 256), dim3(256), 0, stream>>>(Wout, woutb);
  }

  // 2) sequential GRU: single workgroup, W_hh register-resident, bf16 WMMA
  k_gru<<<dim3(1), dim3(512), 0, stream>>>(xp, Whh, bhh, hseq);

  // 3) output projection GEMM (33.5 GFLOP) + bias + symbolic mask
  const dim3 grid(V_ / 16 / 8, (B_ * S_) / 16 / 8);
  if (use_bf16_w) {
    k_logits<true><<<grid, dim3(256), 0, stream>>>(hseq, (const void*)woutb,
                                                   bout, x, out);
  } else {
    k_logits<false><<<grid, dim3(256), 0, stream>>>(hseq, (const void*)Wout,
                                                    bout, x, out);
  }
}
